// HDCAM_46127948759474
// MI455X (gfx1250) — matrix operand order repaired and checked by reading, not compiled, not run on hardware
//
#include <hip/hip_runtime.h>
#include <hip/hip_bf16.h>
#include <math.h>
#include <stdint.h>

// Problem constants (match reference)
#define B_   8
#define T_   1024
#define V_   256
#define D_   4096
#define K_   8192
#define H_   512
#define M_   (B_ * T_)      // 8192 rows
#define C3H_ (3 * H_)       // 1536
#define DECAYF 0.95f

typedef __bf16 bf16_t;
typedef __attribute__((ext_vector_type(16))) __bf16 v16bf;
typedef __attribute__((ext_vector_type(8)))  __bf16 v8bf;
typedef __attribute__((ext_vector_type(8)))  float  v8f;
typedef unsigned int u32x4 __attribute__((ext_vector_type(4)));
typedef int          i32x4 __attribute__((ext_vector_type(4)));
typedef int          i32x8 __attribute__((ext_vector_type(8)));

// Tensor Data Mover availability (this toolchain: 6-arg clang-23 form)
#if defined(__has_builtin)
#if __has_builtin(__builtin_amdgcn_tensor_load_to_lds)
#define HAVE_TDM 1
#endif
#endif
#ifndef HAVE_TDM
#define HAVE_TDM 0
#endif

// GEMM staging geometry: B tile = 64 N-rows x 128 K bf16 per stage.
// LDS row stride 272B (=68 DWORDs): 68 mod 64 = 4 -> 4-bank skew per row,
// so 16-lane x 16B fragment reads are bank-conflict-free.  The TDM produces
// this layout natively via pad_enable (16B pad per 256B row).
#define KSTAGE 128
#define ROWSTR (KSTAGE + 8)

// ---------------------------------------------------------------------------
// 1) Fused trigram-HDC encode + causal exponential-decay scan.
//    ctx[t] = (hdc[t] + 0.95*acc[t-1]) / norm[t]  == normalized decay matmul.
// ---------------------------------------------------------------------------
__global__ __launch_bounds__(256) void encode_scan_kernel(
    const int* __restrict__ idx, const float* __restrict__ cb,
    bf16_t* __restrict__ ctx)
{
    int b = blockIdx.x >> 4;                         // D/256 = 16 blocks per batch
    int d = ((blockIdx.x & 15) << 8) + threadIdx.x;  // 0..4095
    const int* idxrow = idx + b * T_;
    int dm1 = (d + D_ - 1) & (D_ - 1);
    int dm2 = (d + D_ - 2) & (D_ - 1);
    float acc = 0.f, norm = 0.f;
    int v0 = 0, v1 = 0;                              // idx[t-1], idx[t-2]
    bf16_t* out = ctx + (size_t)b * T_ * D_ + d;
    for (int t = 0; t < T_; ++t) {
        int v = idxrow[t];
        float c = cb[(size_t)v * D_ + d];
        float h = 0.3f * c;
        if (t >= 2) {                                // uniform branch
            float c1 = cb[(size_t)v0 * D_ + dm1];
            float c2 = cb[(size_t)v1 * D_ + dm2];
            h += 0.7f * c * c1 * c2;
        }
        acc  = h + DECAYF * acc;
        norm = 1.f + DECAYF * norm;
        out[(size_t)t * D_] = (bf16_t)(acc / norm);
        v1 = v0; v0 = v;
    }
}

// ---------------------------------------------------------------------------
// 2) fp32 -> bf16 conversions (plain + transposing into NT layout)
// ---------------------------------------------------------------------------
__global__ __launch_bounds__(256) void conv_bf16_kernel(
    const float* __restrict__ src, bf16_t* __restrict__ dst, size_t n)
{
    size_t i = ((size_t)blockIdx.x * 256 + threadIdx.x) * 4;
    if (i + 3 < n) {
        float4 v = *(const float4*)(src + i);
        dst[i + 0] = (bf16_t)v.x; dst[i + 1] = (bf16_t)v.y;
        dst[i + 2] = (bf16_t)v.z; dst[i + 3] = (bf16_t)v.w;
    } else {
        for (size_t j = i; j < n; ++j) dst[j] = (bf16_t)src[j];
    }
}

__global__ __launch_bounds__(256) void conv_transpose_bf16_kernel(
    const float* __restrict__ src, bf16_t* __restrict__ dst, int rows, int cols)
{
    size_t i = (size_t)blockIdx.x * 256 + threadIdx.x;
    size_t total = (size_t)rows * cols;
    if (i >= total) return;
    int c = (int)(i / rows);
    int r = (int)(i % rows);
    dst[(size_t)c * rows + r] = (bf16_t)src[(size_t)r * cols + c];
}

// ---------------------------------------------------------------------------
// WMMA A fragment loader (16x32 bf16, wave32 layout per ISA 7.12.2):
//  lane m = lane&15; lanes<16 hold K 0..7 | 16..23; lanes>=16 K 8..15 | 24..31
//  -> two contiguous 16B global_load_b128.
// ---------------------------------------------------------------------------
__device__ __forceinline__ v16bf load_a_frag(const bf16_t* __restrict__ A,
                                             int lda, int m0, int k0, int lane)
{
    int m  = m0 + (lane & 15);
    int ko = k0 + ((lane >> 4) << 3);
    const v8bf* p = (const v8bf*)(A + (size_t)m * lda + ko);
    v8bf lo = p[0];
    v8bf hi = p[2];           // +16 elements
    v16bf f;
#pragma unroll
    for (int j = 0; j < 8; ++j) { f[j] = lo[j]; f[j + 8] = hi[j]; }
    return f;
}

#if HAVE_TDM
// ---------------------------------------------------------------------------
// Tensor DMA: load one B tile (64 rows x KSTAGE bf16, row stride Kd elems)
// from global into LDS at byte offset lds_off, padding 16B per 256B row.
// D# layout per CDNA5 ISA 8.3/8.4 (2D tensor: groups 2/3 zero).
// ---------------------------------------------------------------------------
__device__ __forceinline__ void tdm_load_b_tile(const bf16_t* gsrc,
                                                unsigned lds_off,
                                                int Kd, int N)
{
    unsigned long long ga = (unsigned long long)(uintptr_t)gsrc;
    u32x4 g0;
    g0[0] = 1u;                                          // count=1 (valid user D#)
    g0[1] = lds_off;                                     // lds_addr
    g0[2] = (unsigned)ga;                                // global_addr[31:0]
    g0[3] = (unsigned)((ga >> 32) & 0x01ffffffu)         // global_addr[56:32]
          | (2u << 30);                                  // type = 2 ("image")
    i32x8 g1;
    g1[0] = (int)((1u << 16)      // data_size = 1 -> 2 bytes
                | (1u << 20)      // pad_enable
                | (5u << 22)      // pad_interval: 64 DWORDs (256B)
                | (3u << 25));    // pad_amount:   4 DWORDs (16B)
    g1[1] = (int)(((unsigned)Kd & 0xffffu) << 16);       // tensor_dim0[15:0]
    g1[2] = (int)((((unsigned)Kd >> 16) & 0xffffu)       // tensor_dim0[31:16]
                | (((unsigned)N & 0xffffu) << 16));      // tensor_dim1[15:0]
    g1[3] = (int)((((unsigned)N >> 16) & 0xffffu)        // tensor_dim1[31:16]
                | ((unsigned)KSTAGE << 16));             // tile_dim0 = 128
    g1[4] = 64;                                          // tile_dim1 = 64 rows
    g1[5] = Kd;                                          // tensor_dim0_stride lo32
    g1[6] = 0;                                           // stride hi | dim1_stride lo
    g1[7] = 0;
    i32x4 z4 = {0, 0, 0, 0};
    i32x8 z8 = {0, 0, 0, 0, 0, 0, 0, 0};
    // 6-arg form on this toolchain: (g0, g1, g2, g3, pad, cpol)
    __builtin_amdgcn_tensor_load_to_lds(g0, g1, z4, z4, z8, 0);
}
#endif

// ---------------------------------------------------------------------------
// 3) NT bf16 GEMM: C(M,N) = A(M,Kd) * Bt(N,Kd)^T, B tile staged in LDS
//    (TDM double-buffered when available), A direct from global.
//    Block = 256 thr = 8 waves; block tile 128x64; wave tile 32x32 (2x2 WMMA).
//    EP 0: bf16 out *alpha | EP 1: bf16 gelu(x+bias) | EP 2: f32 x+bias
// ---------------------------------------------------------------------------
template <int EP>
__device__ __forceinline__ void store_tile(void* Cout, int N,
                                           const float* __restrict__ bias,
                                           float alpha, int ms, int ns,
                                           int rb, int col, const v8f& acc)
{
    int c = ns + col;
    float bv = (EP == 0) ? 0.f : (bias ? bias[c] : 0.f);
#pragma unroll
    for (int r = 0; r < 8; ++r) {
        int row = ms + rb + r;
        float v = acc[r];
        if (EP == 0) {
            ((bf16_t*)Cout)[(size_t)row * N + c] = (bf16_t)(v * alpha);
        } else if (EP == 1) {
            float x = v + bv;
            float g = 0.5f * x * (1.0f + erff(x * 0.70710678118f)); // exact gelu
            ((bf16_t*)Cout)[(size_t)row * N + c] = (bf16_t)g;
        } else {
            ((float*)Cout)[(size_t)row * N + c] = v + bv;
        }
    }
}

template <int EP>
__global__ __launch_bounds__(256) void gemm_nt_kernel(
    const bf16_t* __restrict__ A, const bf16_t* __restrict__ Bt,
    const float* __restrict__ bias, void* __restrict__ Cout,
    int Kd, int N, float alpha)
{
    __shared__ bf16_t bstage[2][64 * ROWSTR];            // 2 x 17KB
    int lane = threadIdx.x & 31;
    int wave = threadIdx.x >> 5;
    int wm = wave & 3, wn = wave >> 2;
    int m0 = blockIdx.x * 128 + wm * 32;
    int nblock = blockIdx.y * 64;
    int n0l = wn * 32;                                   // LDS-local column base
    const bf16_t* btile = Bt + (size_t)nblock * Kd;
    int nst = Kd / KSTAGE;

    v8f acc00 = {}, acc01 = {}, acc10 = {}, acc11 = {};

#if HAVE_TDM
    if (threadIdx.x < 32)                                // one DMA per block
        tdm_load_b_tile(btile, (unsigned)(uintptr_t)&bstage[0][0], Kd, N);
#endif

    for (int s = 0; s < nst; ++s) {
        int cur = s & 1;
#if HAVE_TDM
        if (threadIdx.x < 32) {
            if (s + 1 < nst) {
                tdm_load_b_tile(btile + (size_t)(s + 1) * KSTAGE,
                                (unsigned)(uintptr_t)&bstage[cur ^ 1][0], Kd, N);
                __builtin_amdgcn_s_wait_tensorcnt(1);    // stage s complete
            } else {
                __builtin_amdgcn_s_wait_tensorcnt(0);
            }
        }
        __syncthreads();                                 // LDS tile visible to all
#else
        cur = 0;
        __syncthreads();
        {
            int c0 = threadIdx.x * 4;                    // 4 x 16B per thread
#pragma unroll
            for (int c = c0; c < c0 + 4; ++c) {
                int row = c >> 4, kc = (c & 15) << 3;
                *(v8bf*)&bstage[0][row * ROWSTR + kc] =
                    *(const v8bf*)(btile + (size_t)row * Kd + s * KSTAGE + kc);
            }
        }
        __syncthreads();
#endif
        int kg = s * KSTAGE;
#pragma unroll
        for (int ks = 0; ks < KSTAGE; ks += 32) {
            v16bf a0 = load_a_frag(A, Kd, m0,      kg + ks, lane);
            v16bf a1 = load_a_frag(A, Kd, m0 + 16, kg + ks, lane);
            // B fragments from LDS: lane col n = lane&15, K = 8*(lane>=16)+j | +16
            v16bf b0, b1;
            {
                int nn = n0l + (lane & 15);
                int ko = ks + ((lane >> 4) << 3);
                const v8bf* p0 = (const v8bf*)&bstage[cur][nn * ROWSTR + ko];
                const v8bf* p1 = (const v8bf*)&bstage[cur][(nn + 16) * ROWSTR + ko];
                v8bf l0 = p0[0], h0 = p0[2];
                v8bf l1 = p1[0], h1 = p1[2];
#pragma unroll
                for (int j = 0; j < 8; ++j) {
                    b0[j] = l0[j]; b0[j + 8] = h0[j];
                    b1[j] = l1[j]; b1[j + 8] = h1[j];
                }
            }
            acc00 = __builtin_amdgcn_wmma_f32_16x16x32_bf16(false, a0, false, b0,
                                                            (short)0, acc00, false, false);
            acc01 = __builtin_amdgcn_wmma_f32_16x16x32_bf16(false, a0, false, b1,
                                                            (short)0, acc01, false, false);
            acc10 = __builtin_amdgcn_wmma_f32_16x16x32_bf16(false, a1, false, b0,
                                                            (short)0, acc10, false, false);
            acc11 = __builtin_amdgcn_wmma_f32_16x16x32_bf16(false, a1, false, b1,
                                                            (short)0, acc11, false, false);
        }
        __syncthreads();                                 // done reading bstage[cur]
    }

    int col = lane & 15;
    int rb  = (lane >> 4) << 3;   // C/D layout: VGPR r -> M = r + 8*(lane>=16)
    int n0 = nblock + n0l;
    store_tile<EP>(Cout, N, bias, alpha, m0,      n0,      rb, col, acc00);
    store_tile<EP>(Cout, N, bias, alpha, m0,      n0 + 16, rb, col, acc01);
    store_tile<EP>(Cout, N, bias, alpha, m0 + 16, n0,      rb, col, acc10);
    store_tile<EP>(Cout, N, bias, alpha, m0 + 16, n0 + 16, rb, col, acc11);
}

// ---------------------------------------------------------------------------
// 4) Row softmax over K=8192 (in-place bf16), one block per row.
// ---------------------------------------------------------------------------
__global__ __launch_bounds__(256) void softmax_rows_kernel(bf16_t* __restrict__ s)
{
    __shared__ float red[256];
    bf16_t* p = s + (size_t)blockIdx.x * K_;
    float vals[32];
    float mx = -1e30f;
#pragma unroll
    for (int j = 0; j < 32; ++j) {
        float v = (float)p[threadIdx.x + j * 256];
        vals[j] = v;
        mx = fmaxf(mx, v);
    }
    red[threadIdx.x] = mx; __syncthreads();
    for (int st = 128; st > 0; st >>= 1) {
        if (threadIdx.x < st) red[threadIdx.x] = fmaxf(red[threadIdx.x], red[threadIdx.x + st]);
        __syncthreads();
    }
    mx = red[0]; __syncthreads();
    float sum = 0.f;
#pragma unroll
    for (int j = 0; j < 32; ++j) { vals[j] = expf(vals[j] - mx); sum += vals[j]; }
    red[threadIdx.x] = sum; __syncthreads();
    for (int st = 128; st > 0; st >>= 1) {
        if (threadIdx.x < st) red[threadIdx.x] += red[threadIdx.x + st];
        __syncthreads();
    }
    float inv = 1.0f / red[0];
#pragma unroll
    for (int j = 0; j < 32; ++j)
        p[threadIdx.x + j * 256] = (bf16_t)(vals[j] * inv);
}

// ---------------------------------------------------------------------------
// 5) Concat(hdc_float, retrieved, residual) + LayerNorm over 3H=1536 -> bf16.
// ---------------------------------------------------------------------------
__global__ __launch_bounds__(256) void concat_ln_kernel(
    const float* __restrict__ hdcf, const float* __restrict__ retr,
    const int* __restrict__ idx, const float* __restrict__ res_embed,
    const float* __restrict__ pos_embed, const float* __restrict__ ln_g,
    const float* __restrict__ ln_b, bf16_t* __restrict__ xout)
{
    __shared__ float red[256];
    int row = blockIdx.x;
    int b = row >> 10, t = row & 1023;
    int tok = idx[b * T_ + t];
    float x[6];
    float s = 0.f;
#pragma unroll
    for (int j = 0; j < 6; ++j) {
        int i = threadIdx.x + j * 256;     // branch uniform per j
        float v;
        if (i < H_)            v = hdcf[(size_t)row * H_ + i];
        else if (i < 2 * H_)   v = retr[(size_t)row * H_ + (i - H_)];
        else                   v = res_embed[(size_t)tok * H_ + (i - 2 * H_)]
                                 + pos_embed[(size_t)t * H_ + (i - 2 * H_)];
        x[j] = v; s += v;
    }
    red[threadIdx.x] = s; __syncthreads();
    for (int st = 128; st > 0; st >>= 1) {
        if (threadIdx.x < st) red[threadIdx.x] += red[threadIdx.x + st];
        __syncthreads();
    }
    float mu = red[0] * (1.0f / (float)C3H_);
    __syncthreads();
    float s2 = 0.f;
#pragma unroll
    for (int j = 0; j < 6; ++j) { float d = x[j] - mu; s2 += d * d; }
    red[threadIdx.x] = s2; __syncthreads();
    for (int st = 128; st > 0; st >>= 1) {
        if (threadIdx.x < st) red[threadIdx.x] += red[threadIdx.x + st];
        __syncthreads();
    }
    float rstd = rsqrtf(red[0] * (1.0f / (float)C3H_) + 1e-5f);
#pragma unroll
    for (int j = 0; j < 6; ++j) {
        int i = threadIdx.x + j * 256;
        float v = (x[j] - mu) * rstd * ln_g[i] + ln_b[i];
        xout[(size_t)row * C3H_ + i] = (bf16_t)v;
    }
}

// ---------------------------------------------------------------------------
// Host-side pipeline (all on `stream`; graph-capture safe)
// ---------------------------------------------------------------------------
extern "C" void kernel_launch(void* const* d_in, const int* in_sizes, int n_in,
                              void* d_out, int out_size, void* d_ws, size_t ws_size,
                              hipStream_t stream)
{
    const int*   idx   = (const int*)  d_in[0];
    const float* cb    = (const float*)d_in[1];
    const float* keys  = (const float*)d_in[2];
    const float* vals  = (const float*)d_in[3];
    const float* projw = (const float*)d_in[4];
    const float* projb = (const float*)d_in[5];
    const float* rese  = (const float*)d_in[6];
    const float* pose  = (const float*)d_in[7];
    const float* lng   = (const float*)d_in[8];
    const float* lnb   = (const float*)d_in[9];
    const float* w1    = (const float*)d_in[10];
    const float* b1    = (const float*)d_in[11];
    const float* w2    = (const float*)d_in[12];
    const float* b2    = (const float*)d_in[13];
    const float* w3    = (const float*)d_in[14];
    const float* b3    = (const float*)d_in[15];
    const float* hw    = (const float*)d_in[16];
    const float* hb    = (const float*)d_in[17];

    char* ws = (char*)d_ws;
    size_t off = 0;
    auto alloc = [&](size_t bytes) -> void* {
        void* p = ws + off;
        off += (bytes + 255) & ~(size_t)255;
        return p;
    };
    bf16_t* ctx    = (bf16_t*)alloc((size_t)M_ * D_  * 2);   //  64 MB
    bf16_t* keysb  = (bf16_t*)alloc((size_t)K_ * D_  * 2);   //  64 MB
    bf16_t* valsT  = (bf16_t*)alloc((size_t)H_ * K_  * 2);   //   8 MB
    bf16_t* projT  = (bf16_t*)alloc((size_t)H_ * D_  * 2);   //   4 MB
    bf16_t* w1T    = (bf16_t*)alloc((size_t)H_ * C3H_* 2);
    bf16_t* w2T    = (bf16_t*)alloc((size_t)H_ * H_  * 2);
    bf16_t* w3T    = (bf16_t*)alloc((size_t)H_ * H_  * 2);
    bf16_t* headT  = (bf16_t*)alloc((size_t)V_ * H_  * 2);
    bf16_t* scores = (bf16_t*)alloc((size_t)M_ * K_  * 2);   // 128 MB (probs in place)
    float*  retr   = (float*) alloc((size_t)M_ * H_  * 4);   //  16 MB
    float*  hdcf   = (float*) alloc((size_t)M_ * H_  * 4);   //  16 MB
    bf16_t* xln    = (bf16_t*)alloc((size_t)M_ * C3H_* 2);   //  24 MB
    bf16_t* h1     = (bf16_t*)alloc((size_t)M_ * H_  * 2);
    bf16_t* h2     = (bf16_t*)alloc((size_t)M_ * H_  * 2);
    bf16_t* h3     = (bf16_t*)alloc((size_t)M_ * H_  * 2);
    (void)ws_size; (void)in_sizes; (void)n_in; (void)out_size;

    // --- precision/layout conversions -------------------------------------
    {
        size_t n = (size_t)K_ * D_;
        conv_bf16_kernel<<<(unsigned)((n / 4 + 255) / 256), 256, 0, stream>>>(keys, keysb, n);
    }
    conv_transpose_bf16_kernel<<<(unsigned)(((size_t)K_ * H_   + 255) / 256), 256, 0, stream>>>(vals,  valsT, K_,   H_);
    conv_transpose_bf16_kernel<<<(unsigned)(((size_t)D_ * H_   + 255) / 256), 256, 0, stream>>>(projw, projT, D_,   H_);
    conv_transpose_bf16_kernel<<<(unsigned)(((size_t)C3H_ * H_ + 255) / 256), 256, 0, stream>>>(w1,    w1T,   C3H_, H_);
    conv_transpose_bf16_kernel<<<(unsigned)(((size_t)H_ * H_   + 255) / 256), 256, 0, stream>>>(w2,    w2T,   H_,   H_);
    conv_transpose_bf16_kernel<<<(unsigned)(((size_t)H_ * H_   + 255) / 256), 256, 0, stream>>>(w3,    w3T,   H_,   H_);
    conv_transpose_bf16_kernel<<<(unsigned)(((size_t)H_ * V_   + 255) / 256), 256, 0, stream>>>(hw,    headT, H_,   V_);

    // --- 1. HDC encode + decay scan -> ctx (bf16) -------------------------
    encode_scan_kernel<<<B_ * (D_ / 256), 256, 0, stream>>>(idx, cb, ctx);

    // --- 2. scores = ctx @ keys^T * D^-0.5 ; softmax in place -------------
    gemm_nt_kernel<0><<<dim3(M_ / 128, K_ / 64), 256, 0, stream>>>(
        ctx, keysb, nullptr, scores, D_, K_, 0.015625f /* 4096^-0.5 */);
    softmax_rows_kernel<<<M_, 256, 0, stream>>>(scores);

    // --- 3. retrieved = probs @ values ; hdc_float = ctx @ proj_w + b ----
    gemm_nt_kernel<2><<<dim3(M_ / 128, H_ / 64), 256, 0, stream>>>(
        scores, valsT, nullptr, retr, K_, H_, 1.f);
    gemm_nt_kernel<2><<<dim3(M_ / 128, H_ / 64), 256, 0, stream>>>(
        ctx, projT, projb, hdcf, D_, H_, 1.f);

    // --- 4. concat + layernorm -> bf16 x ----------------------------------
    concat_ln_kernel<<<M_, 256, 0, stream>>>(hdcf, retr, idx, rese, pose,
                                             lng, lnb, xln);

    // --- 5. MLP (gelu epilogues) + head -----------------------------------
    gemm_nt_kernel<1><<<dim3(M_ / 128, H_ / 64), 256, 0, stream>>>(
        xln, w1T, b1, h1, C3H_, H_, 1.f);
    gemm_nt_kernel<1><<<dim3(M_ / 128, H_ / 64), 256, 0, stream>>>(
        h1, w2T, b2, h2, H_, H_, 1.f);
    gemm_nt_kernel<1><<<dim3(M_ / 128, H_ / 64), 256, 0, stream>>>(
        h2, w3T, b3, h3, H_, H_, 1.f);
    gemm_nt_kernel<2><<<dim3(M_ / 128, V_ / 64), 256, 0, stream>>>(
        h3, headT, hb, (float*)d_out, H_, V_, 1.f);
}
